// FastestFlashMLA_738734374972
// MI455X (gfx1250) — compile-verified, matrix operand order checked by reading, not run on hardware
//
#include <hip/hip_runtime.h>
#include <hip/hip_bf16.h>

#define B_   2
#define S_   2048
#define HID_ 2048
#define H_   16
#define HD_  128
#define RD_  64
#define LD_  512
#define AD_  192
#define BS_  (B_*S_)
#define SCALE_ 0.07216878364870322f   /* 1/sqrt(192) */
#define LN10000_ 9.210340371976184f

typedef __bf16 bf16;
typedef __attribute__((ext_vector_type(16))) __bf16 v16bf;
typedef __attribute__((ext_vector_type(8)))  __bf16 v8bf;
typedef __attribute__((ext_vector_type(8)))  float  v8f;

#define CDIV(a,b) (((a)+(b)-1)/(b))

// Load one WMMA bf16 operand fragment: 8 halves at p, 8 halves at p+16.
// Matches the 16-bit A/B 16x32 VGPR layout: lane l holds row/col (l&15),
// K = (l>=16?8:0) + {0..7, 16..23}.
__device__ __forceinline__ v16bf frag_ld(const bf16* p) {
    v8bf lo = *(const v8bf*)p;
    v8bf hi = *(const v8bf*)(p + 16);
    return __builtin_shufflevector(lo, hi, 0,1,2,3,4,5,6,7,8,9,10,11,12,13,14,15);
}

// Async copy 16 bytes global -> LDS (per lane), tracked by ASYNCcnt.
// lds_off is the wave-relative LDS byte address (= low 32 bits of the
// generic pointer to a __shared__ object, per the CDNA5 aperture layout).
__device__ __forceinline__ void async_cp16(unsigned lds_off, const void* g) {
    unsigned long long ga = (unsigned long long)(uintptr_t)g;
    asm volatile("global_load_async_to_lds_b128 %0, %1, off"
                 :: "v"(lds_off), "v"(ga) : "memory");
}
__device__ __forceinline__ void wait_async0() {
    asm volatile("s_wait_asynccnt 0x0" ::: "memory");
}

// ---------------- conversion / transpose kernels ----------------

__global__ void k_cvt(const float* __restrict__ src, bf16* __restrict__ dst, int n) {
    int i = blockIdx.x * 256 + threadIdx.x;
    if (i < n) dst[i] = (bf16)src[i];
}

// src [K,N] f32 row-major -> dst [N,K] bf16 row-major (transpose)
__global__ void k_cvtT(const float* __restrict__ src, bf16* __restrict__ dst, int K, int N) {
    int i = blockIdx.x * 256 + threadIdx.x;
    if (i >= K * N) return;
    int k = i / N, n = i % N;
    dst[(size_t)n * K + k] = (bf16)src[i];
}

// wo [H*AD, HID] -> woT [HID, H*HD], gathering only the first HD rows of each
// head's AD block (the padded RD rows multiply zeros in the reference).
__global__ void k_cvtWo(const float* __restrict__ wo, bf16* __restrict__ dst) {
    int i = blockIdx.x * 256 + threadIdx.x;
    if (i >= HID_ * H_ * HD_) return;
    int n = i / (H_ * HD_);
    int c = i % (H_ * HD_);
    int h = c / HD_, d = c % HD_;
    dst[i] = (bf16)wo[(size_t)(h * AD_ + d) * HID_ + n];
}

// ---------- small direct GEMM (for N=64): C[M,N] = A[M,K] * Bt[N,K]^T ------

__global__ __launch_bounds__(128) void gemm_bf16_small(const bf16* __restrict__ A,
                                                       const bf16* __restrict__ Bt,
                                                       void* __restrict__ Cp,
                                                       int M, int N, int K, int c_is_f32) {
    int lane = threadIdx.x & 31;
    int wave = threadIdx.x >> 5;
    int half = lane >> 4;
    int ln   = lane & 15;
    int kb   = half * 8;
    int row0 = blockIdx.y * 64 + wave * 16;
    int col0 = blockIdx.x * 64;
    v8f acc[4] = {};
    const bf16* arow = A + (size_t)(row0 + ln) * K + kb;
    for (int k0 = 0; k0 < K; k0 += 32) {
        v16bf a = frag_ld(arow + k0);
#pragma unroll
        for (int t = 0; t < 4; ++t) {
            const bf16* brow = Bt + (size_t)(col0 + t * 16 + ln) * K + kb + k0;
            v16bf b = frag_ld(brow);
            acc[t] = __builtin_amdgcn_wmma_f32_16x16x32_bf16(false, a, false, b,
                                                             (short)0, acc[t], false, false);
        }
    }
#pragma unroll
    for (int t = 0; t < 4; ++t)
#pragma unroll
        for (int j = 0; j < 8; ++j) {
            int m = j + half * 8;
            size_t idx = (size_t)(row0 + m) * N + col0 + t * 16 + ln;
            if (c_is_f32) ((float*)Cp)[idx] = acc[t][j];
            else          ((bf16*)Cp)[idx]  = (bf16)acc[t][j];
        }
}

// ---------- main GEMM: 128x128 block tile, async-LDS double buffered -------
// 256 threads = 8 waves in a 4(m) x 2(n) grid; each wave computes 32x64.
// K staged in chunks of 64 via global_load_async_to_lds_b128.

#define TM 128
#define TN 128
#define TK 64

__global__ __launch_bounds__(256) void gemm_bf16_lds(const bf16* __restrict__ A,
                                                     const bf16* __restrict__ Bt,
                                                     void* __restrict__ Cp,
                                                     int M, int N, int K, int c_is_f32) {
    __shared__ alignas(16) bf16 Ash[2][TM][TK];
    __shared__ alignas(16) bf16 Bsh[2][TN][TK];
    int tid  = threadIdx.x;
    int lane = tid & 31, wave = tid >> 5;
    int wm = wave & 3, wn = wave >> 2;
    int half = lane >> 4, ln = lane & 15;
    int kb = half * 8;
    int row0 = blockIdx.y * TM;
    int col0 = blockIdx.x * TN;

    unsigned ash0 = (unsigned)(uintptr_t)&Ash[0][0][0];
    unsigned bsh0 = (unsigned)(uintptr_t)&Bsh[0][0][0];

    // per-thread copy slices: 1024 16B segments per tile, 4 per thread
    int crow[4], cseg[4];
#pragma unroll
    for (int t = 0; t < 4; ++t) {
        int idx = tid + t * 256;
        crow[t] = idx >> 3;
        cseg[t] = (idx & 7) * 8;
    }

    v8f acc[2][4] = {};
    int nk = K / TK;

    // prologue: stage chunk 0 into buffer 0
#pragma unroll
    for (int t = 0; t < 4; ++t)
        async_cp16(ash0 + (unsigned)(crow[t] * TK + cseg[t]) * 2,
                   A + (size_t)(row0 + crow[t]) * K + cseg[t]);
#pragma unroll
    for (int t = 0; t < 4; ++t)
        async_cp16(bsh0 + (unsigned)(crow[t] * TK + cseg[t]) * 2,
                   Bt + (size_t)(col0 + crow[t]) * K + cseg[t]);

    for (int i = 0; i < nk; ++i) {
        wait_async0();
        __syncthreads();
        int cur = i & 1;
        if (i + 1 < nk) {
            int nxt = 1 - cur;
            int k0 = (i + 1) * TK;
            unsigned abase = ash0 + (unsigned)(nxt * TM * TK * 2);
            unsigned bbase = bsh0 + (unsigned)(nxt * TN * TK * 2);
#pragma unroll
            for (int t = 0; t < 4; ++t)
                async_cp16(abase + (unsigned)(crow[t] * TK + cseg[t]) * 2,
                           A + (size_t)(row0 + crow[t]) * K + k0 + cseg[t]);
#pragma unroll
            for (int t = 0; t < 4; ++t)
                async_cp16(bbase + (unsigned)(crow[t] * TK + cseg[t]) * 2,
                           Bt + (size_t)(col0 + crow[t]) * K + k0 + cseg[t]);
        }
#pragma unroll
        for (int ks = 0; ks < TK; ks += 32) {
            v16bf af[2], bfv[4];
#pragma unroll
            for (int r = 0; r < 2; ++r)
                af[r] = frag_ld(&Ash[cur][wm * 32 + r * 16 + ln][ks + kb]);
#pragma unroll
            for (int t = 0; t < 4; ++t)
                bfv[t] = frag_ld(&Bsh[cur][wn * 64 + t * 16 + ln][ks + kb]);
#pragma unroll
            for (int r = 0; r < 2; ++r)
#pragma unroll
                for (int t = 0; t < 4; ++t)
                    acc[r][t] = __builtin_amdgcn_wmma_f32_16x16x32_bf16(false, af[r], false, bfv[t],
                                                                        (short)0, acc[r][t], false, false);
        }
    }

#pragma unroll
    for (int r = 0; r < 2; ++r)
#pragma unroll
        for (int t = 0; t < 4; ++t)
#pragma unroll
            for (int j = 0; j < 8; ++j) {
                int m = wm * 32 + r * 16 + j + half * 8;
                size_t idx = (size_t)(row0 + m) * N + col0 + wn * 64 + t * 16 + ln;
                if (c_is_f32) ((float*)Cp)[idx] = acc[r][t][j];
                else          ((bf16*)Cp)[idx]  = (bf16)acc[r][t][j];
            }
}

// ---------------- assemble Q/K/V (+RoPE) ----------------

__global__ void k_asm_qn(const bf16* __restrict__ t_qn, bf16* __restrict__ Q) {
    int i = blockIdx.x * 256 + threadIdx.x;
    if (i >= BS_ * H_ * HD_) return;
    int d = i % HD_; int t = i / HD_; int h = t % H_; int bs = t / H_;
    int b = bs / S_, s = bs % S_;
    Q[((size_t)(b * H_ + h) * S_ + s) * AD_ + d] =
        t_qn[(size_t)bs * (H_ * HD_) + h * HD_ + d];
}

__global__ void k_asm_qrope(const bf16* __restrict__ t_qr, bf16* __restrict__ Q) {
    int i = blockIdx.x * 256 + threadIdx.x;
    if (i >= BS_ * H_ * (RD_ / 2)) return;
    int j = i % (RD_ / 2); int t = i / (RD_ / 2); int h = t % H_; int bs = t / H_;
    int b = bs / S_, s = bs % S_;
    float x1 = (float)t_qr[(size_t)bs * (H_ * RD_) + h * RD_ + j];
    float x2 = (float)t_qr[(size_t)bs * (H_ * RD_) + h * RD_ + RD_ / 2 + j];
    float freq = (float)s * __expf(-(2.0f * j / RD_) * LN10000_);
    float sn, cs; __sincosf(freq, &sn, &cs);
    size_t base = ((size_t)(b * H_ + h) * S_ + s) * AD_ + HD_;
    Q[base + j]           = (bf16)(x1 * cs - x2 * sn);
    Q[base + RD_ / 2 + j] = (bf16)(x2 * cs + x1 * sn);
}

__global__ void k_asm_kn(const bf16* __restrict__ t_kv, bf16* __restrict__ Kb) {
    int i = blockIdx.x * 256 + threadIdx.x;
    if (i >= BS_ * H_ * HD_) return;
    int d = i % HD_; int t = i / HD_; int h = t % H_; int bs = t / H_;
    int b = bs / S_, s = bs % S_;
    Kb[((size_t)(b * H_ + h) * S_ + s) * AD_ + d] =
        t_kv[(size_t)bs * (H_ * 2 * HD_) + h * 2 * HD_ + d];
}

__global__ void k_asm_krope(const bf16* __restrict__ t_kr, bf16* __restrict__ Kb) {
    int i = blockIdx.x * 256 + threadIdx.x;
    if (i >= BS_ * H_ * (RD_ / 2)) return;
    int j = i % (RD_ / 2); int t = i / (RD_ / 2); int h = t % H_; int bs = t / H_;
    int b = bs / S_, s = bs % S_;
    float x1 = (float)t_kr[(size_t)bs * RD_ + j];
    float x2 = (float)t_kr[(size_t)bs * RD_ + RD_ / 2 + j];
    float freq = (float)s * __expf(-(2.0f * j / RD_) * LN10000_);
    float sn, cs; __sincosf(freq, &sn, &cs);
    size_t base = ((size_t)(b * H_ + h) * S_ + s) * AD_ + HD_;
    Kb[base + j]           = (bf16)(x1 * cs - x2 * sn);
    Kb[base + RD_ / 2 + j] = (bf16)(x2 * cs + x1 * sn);
}

__global__ void k_asm_vt(const bf16* __restrict__ t_kv, bf16* __restrict__ Vt) {
    int i = blockIdx.x * 256 + threadIdx.x;
    if (i >= BS_ * H_ * HD_) return;
    int d = i % HD_; int t = i / HD_; int h = t % H_; int bs = t / H_;
    int b = bs / S_, s = bs % S_;
    Vt[((size_t)(b * H_ + h) * HD_ + d) * S_ + s] =
        t_kv[(size_t)bs * (H_ * 2 * HD_) + h * 2 * HD_ + HD_ + d];
}

// ---------------- flash attention (causal, online softmax) ----------------
// grid: (S/64, B*H), block: 128 (4 waves, 16 q-rows each).
// K/V chunks (32 keys) staged in double-buffered LDS via async copies and
// shared by all 4 waves; P transposed through a per-wave LDS tile.

__global__ __launch_bounds__(128) void k_attn(const bf16* __restrict__ Q,
                                              const bf16* __restrict__ K,
                                              const bf16* __restrict__ Vt,
                                              bf16* __restrict__ attn) {
    __shared__ alignas(16) bf16 Ksh[2][32][AD_];   // 24 KB
    __shared__ alignas(16) bf16 Vsh[2][HD_][32];   // 16 KB
    __shared__ alignas(16) bf16 Pbuf[4][16][32];   //  4 KB
    int tid  = threadIdx.x;
    int lane = tid & 31;
    int wave = tid >> 5;
    int half = lane >> 4, ln = lane & 15;
    int kb = half * 8;
    int bh = blockIdx.y;
    int b = bh / H_, h = bh % H_;
    int qt = blockIdx.x;
    int qrow0 = qt * 64 + wave * 16;

    const bf16* Kbh = K  + (size_t)bh * S_ * AD_;
    const bf16* Vbh = Vt + (size_t)bh * HD_ * S_;
    unsigned ksh0 = (unsigned)(uintptr_t)&Ksh[0][0][0];
    unsigned vsh0 = (unsigned)(uintptr_t)&Vsh[0][0][0];

    // stage one 32-key chunk (K: 32x192, V^T: 128x32) into LDS buffer cbuf
    auto stage = [&](int cbuf, int key0) {
        unsigned kbase = ksh0 + (unsigned)cbuf * (32 * AD_ * 2);
#pragma unroll
        for (int t = 0; t < 6; ++t) {          // 768 segs / 128 threads
            int idx = tid + t * 128;
            int r = idx / 24, sg = (idx % 24) * 8;
            async_cp16(kbase + (unsigned)(r * AD_ + sg) * 2,
                       Kbh + (size_t)(key0 + r) * AD_ + sg);
        }
        unsigned vbase = vsh0 + (unsigned)cbuf * (HD_ * 32 * 2);
#pragma unroll
        for (int t = 0; t < 4; ++t) {          // 512 segs / 128 threads
            int idx = tid + t * 128;
            int d = idx >> 2, sg = (idx & 3) * 8;
            async_cp16(vbase + (unsigned)(d * 32 + sg) * 2,
                       Vbh + (size_t)d * S_ + key0 + sg);
        }
    };

    const bf16* Qbase = Q + ((size_t)bh * S_ + qrow0 + ln) * AD_ + kb;
    v16bf qf[6];
#pragma unroll
    for (int t = 0; t < 6; ++t) qf[t] = frag_ld(Qbase + t * 32);

    v8f o[8] = {};
    float mrow[8], lrow[8];
#pragma unroll
    for (int j = 0; j < 8; ++j) { mrow[j] = -3.0e38f; lrow[j] = 0.0f; }

    int nchunk = 2 * qt + 2;
    stage(0, 0);

    for (int c = 0; c < nchunk; ++c) {
        wait_async0();
        __syncthreads();
        int cur = c & 1;
        if (c + 1 < nchunk) stage(1 - cur, (c + 1) * 32);
        int key0 = c * 32;

        v8f sfrag[2] = {};
#pragma unroll
        for (int nt = 0; nt < 2; ++nt)
#pragma unroll
            for (int t = 0; t < 6; ++t) {
                v16bf kf = frag_ld(&Ksh[cur][nt * 16 + ln][t * 32 + kb]);
                sfrag[nt] = __builtin_amdgcn_wmma_f32_16x16x32_bf16(false, qf[t], false, kf,
                                                                    (short)0, sfrag[nt], false, false);
            }

        float rmax[8];
#pragma unroll
        for (int j = 0; j < 8; ++j) {
            int qg = qrow0 + j + half * 8;
            float s0 = sfrag[0][j] * SCALE_;
            float s1 = sfrag[1][j] * SCALE_;
            if (key0 + ln      > qg) s0 = -3.0e38f;
            if (key0 + 16 + ln > qg) s1 = -3.0e38f;
            sfrag[0][j] = s0; sfrag[1][j] = s1;
            rmax[j] = fmaxf(s0, s1);
        }
#pragma unroll
        for (int j = 0; j < 8; ++j)
#pragma unroll
            for (int off = 1; off < 16; off <<= 1)
                rmax[j] = fmaxf(rmax[j], __shfl_xor(rmax[j], off, 32));

        float escale[8];
#pragma unroll
        for (int j = 0; j < 8; ++j) {
            float mnew = fmaxf(mrow[j], rmax[j]);
            escale[j] = __expf(mrow[j] - mnew);
            mrow[j] = mnew;
        }
        float rsum[8];
#pragma unroll
        for (int j = 0; j < 8; ++j) {
            float p0 = __expf(sfrag[0][j] - mrow[j]);
            float p1 = __expf(sfrag[1][j] - mrow[j]);
            sfrag[0][j] = p0; sfrag[1][j] = p1;
            rsum[j] = p0 + p1;
        }
#pragma unroll
        for (int j = 0; j < 8; ++j) {
#pragma unroll
            for (int off = 1; off < 16; off <<= 1)
                rsum[j] += __shfl_xor(rsum[j], off, 32);
            lrow[j] = lrow[j] * escale[j] + rsum[j];
        }
#pragma unroll
        for (int t = 0; t < 8; ++t)
#pragma unroll
            for (int j = 0; j < 8; ++j)
                o[t][j] *= escale[j];

        // redistribute P (C-layout lane=key) -> A-layout (lane=qrow) via LDS
#pragma unroll
        for (int j = 0; j < 8; ++j) {
            Pbuf[wave][j + half * 8][ln]      = (bf16)sfrag[0][j];
            Pbuf[wave][j + half * 8][16 + ln] = (bf16)sfrag[1][j];
        }
        __syncthreads();
        v16bf pf = frag_ld(&Pbuf[wave][ln][kb]);

#pragma unroll
        for (int t = 0; t < 8; ++t) {
            v16bf vf = frag_ld(&Vsh[cur][t * 16 + ln][kb]);
            o[t] = __builtin_amdgcn_wmma_f32_16x16x32_bf16(false, pf, false, vf,
                                                           (short)0, o[t], false, false);
        }
    }

#pragma unroll
    for (int t = 0; t < 8; ++t)
#pragma unroll
        for (int j = 0; j < 8; ++j) {
            int qg = qrow0 + j + half * 8;
            attn[(size_t)(b * S_ + qg) * (H_ * HD_) + h * HD_ + t * 16 + ln] =
                (bf16)(o[t][j] / lrow[j]);
        }
}

// ---------------- host launch ----------------

extern "C" void kernel_launch(void* const* d_in, const int* in_sizes, int n_in,
                              void* d_out, int out_size, void* d_ws, size_t ws_size,
                              hipStream_t stream) {
    const float* x        = (const float*)d_in[0];
    const float* wq       = (const float*)d_in[1];
    const float* wq_rope  = (const float*)d_in[2];
    const float* wkv_down = (const float*)d_in[3];
    const float* wkv_up   = (const float*)d_in[4];
    const float* wk_rope  = (const float*)d_in[5];
    const float* wo       = (const float*)d_in[6];
    float* out = (float*)d_out;

    char* ws = (char*)d_ws;
    size_t off = 0;
    auto alloc_bf = [&](size_t elems) -> bf16* {
        bf16* p = (bf16*)(ws + off);
        off += ((elems * sizeof(bf16) + 255) & ~(size_t)255);
        return p;
    };
    bf16* xb    = alloc_bf((size_t)BS_ * HID_);
    bf16* wqT   = alloc_bf((size_t)(H_ * HD_) * HID_);
    bf16* wqrT  = alloc_bf((size_t)(H_ * RD_) * HID_);
    bf16* wkdT  = alloc_bf((size_t)LD_ * HID_);
    bf16* wkuT  = alloc_bf((size_t)(H_ * 2 * HD_) * LD_);
    bf16* wkrT  = alloc_bf((size_t)RD_ * HID_);
    bf16* woT   = alloc_bf((size_t)HID_ * (H_ * HD_));
    bf16* t_qn  = alloc_bf((size_t)BS_ * (H_ * HD_));
    bf16* t_qr  = alloc_bf((size_t)BS_ * (H_ * RD_));
    bf16* t_lat = alloc_bf((size_t)BS_ * LD_);
    bf16* t_kv  = alloc_bf((size_t)BS_ * (H_ * 2 * HD_));
    bf16* t_kr  = alloc_bf((size_t)BS_ * RD_);
    bf16* Qb    = alloc_bf((size_t)B_ * H_ * S_ * AD_);
    bf16* Kb    = alloc_bf((size_t)B_ * H_ * S_ * AD_);
    bf16* Vtb   = alloc_bf((size_t)B_ * H_ * HD_ * S_);
    bf16* attnb = alloc_bf((size_t)BS_ * (H_ * HD_));
    (void)ws_size; (void)in_sizes; (void)n_in; (void)out_size;

    // 1) convert inputs / transpose weights to bf16
    k_cvt<<<CDIV(BS_ * HID_, 256), 256, 0, stream>>>(x, xb, BS_ * HID_);
    k_cvtT<<<CDIV(HID_ * H_ * HD_, 256), 256, 0, stream>>>(wq, wqT, HID_, H_ * HD_);
    k_cvtT<<<CDIV(HID_ * H_ * RD_, 256), 256, 0, stream>>>(wq_rope, wqrT, HID_, H_ * RD_);
    k_cvtT<<<CDIV(HID_ * LD_, 256), 256, 0, stream>>>(wkv_down, wkdT, HID_, LD_);
    k_cvtT<<<CDIV(LD_ * H_ * 2 * HD_, 256), 256, 0, stream>>>(wkv_up, wkuT, LD_, H_ * 2 * HD_);
    k_cvtT<<<CDIV(HID_ * RD_, 256), 256, 0, stream>>>(wk_rope, wkrT, HID_, RD_);
    k_cvtWo<<<CDIV(HID_ * H_ * HD_, 256), 256, 0, stream>>>(wo, woT);

    // 2) projection GEMMs (async-LDS path; N=64 one uses the direct kernel)
    gemm_bf16_lds<<<dim3((H_ * HD_) / TN, BS_ / TM), 256, 0, stream>>>(xb, wqT, t_qn,
        BS_, H_ * HD_, HID_, 0);
    gemm_bf16_lds<<<dim3((H_ * RD_) / TN, BS_ / TM), 256, 0, stream>>>(xb, wqrT, t_qr,
        BS_, H_ * RD_, HID_, 0);
    gemm_bf16_lds<<<dim3(LD_ / TN, BS_ / TM), 256, 0, stream>>>(xb, wkdT, t_lat,
        BS_, LD_, HID_, 0);
    gemm_bf16_small<<<dim3(RD_ / 64, BS_ / 64), 128, 0, stream>>>(xb, wkrT, t_kr,
        BS_, RD_, HID_, 0);
    gemm_bf16_lds<<<dim3((H_ * 2 * HD_) / TN, BS_ / TM), 256, 0, stream>>>(t_lat, wkuT, t_kv,
        BS_, H_ * 2 * HD_, LD_, 0);

    // 3) assemble Q/K/V with RoPE
    k_asm_qn<<<CDIV(BS_ * H_ * HD_, 256), 256, 0, stream>>>(t_qn, Qb);
    k_asm_qrope<<<CDIV(BS_ * H_ * (RD_ / 2), 256), 256, 0, stream>>>(t_qr, Qb);
    k_asm_kn<<<CDIV(BS_ * H_ * HD_, 256), 256, 0, stream>>>(t_kv, Kb);
    k_asm_krope<<<CDIV(BS_ * H_ * (RD_ / 2), 256), 256, 0, stream>>>(t_kr, Kb);
    k_asm_vt<<<CDIV(BS_ * H_ * HD_, 256), 256, 0, stream>>>(t_kv, Vtb);

    // 4) causal flash attention (async-staged K/V)
    k_attn<<<dim3(S_ / 64, B_ * H_), 128, 0, stream>>>(Qb, Kb, Vtb, attnb);

    // 5) output projection (f32 store to d_out)
    gemm_bf16_lds<<<dim3(HID_ / TN, BS_ / TM), 256, 0, stream>>>(attnb, woT, out,
        BS_, HID_, H_ * HD_, 1);
}
